// Attention_78563541778641
// MI455X (gfx1250) — matrix operand order repaired and checked by reading, not compiled, not run on hardware
//
#include <hip/hip_runtime.h>

// ---------------------------------------------------------------------------
// CDNA5 (gfx1250) wave32 WMMA types and helpers
// ---------------------------------------------------------------------------
typedef __attribute__((ext_vector_type(16))) __bf16 v16bf;
typedef __attribute__((ext_vector_type(8)))  float  v8f;

// 16-bit A-matrix 16x32 layout (CDNA5 ISA 7.12.2):
// VGPR g (g<4): K = 2g,2g+1 (+8 for lanes 16..31); g>=4: K = 16+2(g-4) (+8)
__device__ __forceinline__ int kmapA(int g, int hi) {
  return ((g < 4) ? (2 * g) : (16 + 2 * (g - 4))) + hi * 8;
}
// 16-bit B-matrix 32x16 layout (same K order as A):
// VGPR g (g<4): K = 2g,2g+1 (+8 for lanes 16..31); g>=4: K = 16+2(g-4) (+8)
__device__ __forceinline__ int kmapB(int g, int hi) {
  return ((g < 4) ? (2 * g) : (16 + 2 * (g - 4))) + hi * 8;
}

__device__ __forceinline__ v8f v8f_zero() {
  v8f z;
#pragma unroll
  for (int i = 0; i < 8; ++i) z[i] = 0.0f;
  return z;
}

__device__ __forceinline__ v8f wmma_bf16(v16bf a, v16bf b, v8f c) {
  // (neg_a, A, neg_b, B, c_mod, C, reuse_a, reuse_b)
  return __builtin_amdgcn_wmma_f32_16x16x32_bf16(false, a, false, b, (short)0, c,
                                                 false, false);
}

template <int W>
__device__ __forceinline__ v16bf load_fragA_lds(const unsigned short (*lds)[W],
                                                int row, int hi, int kofs) {
  union { v16bf v; unsigned int u[8]; } f;
#pragma unroll
  for (int g = 0; g < 8; ++g)
    f.u[g] = *(const unsigned int*)&lds[row][kmapA(g, hi) + kofs];
  return f.v;
}

template <int W>
__device__ __forceinline__ v16bf load_fragB_lds(const unsigned short (*lds)[W],
                                                int row, int hi, int kofs) {
  union { v16bf v; unsigned int u[8]; } f;
#pragma unroll
  for (int g = 0; g < 8; ++g)
    f.u[g] = *(const unsigned int*)&lds[row][kmapB(g, hi) + kofs];
  return f.v;
}

__device__ __forceinline__ v16bf load_fragA_glb(const __bf16* base, int ld,
                                                int row, int hi, int kofs) {
  union { v16bf v; unsigned int u[8]; } f;
#pragma unroll
  for (int g = 0; g < 8; ++g)
    f.u[g] = *(const unsigned int*)(base + (size_t)row * ld + kmapA(g, hi) + kofs);
  return f.v;
}

__device__ __forceinline__ unsigned short bf16_bits(float x) {
  union { __bf16 h; unsigned short s; } u;
  u.h = (__bf16)x;
  return u.s;
}

// LDS byte offset of a shared-memory pointer (aperture puts it in addr[31:0])
__device__ __forceinline__ unsigned lds_off(const void* p) {
  return (unsigned)(unsigned long long)p;
}

// 16B async copy global -> LDS (tracked on ASYNCcnt, no VGPR round-trip)
__device__ __forceinline__ void async_copy_b128(unsigned lds_addr,
                                                const void* gptr) {
  asm volatile("global_load_async_to_lds_b128 %0, %1, off"
               :: "v"(lds_addr), "v"((unsigned long long)gptr) : "memory");
}
__device__ __forceinline__ void wait_async0() {
  asm volatile("s_wait_asynccnt 0" ::: "memory");
}

// ---------------------------------------------------------------------------
// Precision staging: fp32 -> bf16 (copy / transposed copy)
// ---------------------------------------------------------------------------
__global__ __launch_bounds__(256) void k_cast_bf16(const float* __restrict__ s,
                                                   __bf16* __restrict__ d) {
  const int i = blockIdx.x * 256 + threadIdx.x;
  d[i] = (__bf16)s[i];
}

// src [K][N] fp32 -> dst [N][K] bf16 (coalesced writes)
__global__ __launch_bounds__(256) void k_transpose_bf16(const float* __restrict__ s,
                                                        __bf16* __restrict__ d,
                                                        int K, int N) {
  const int i = blockIdx.x * 256 + threadIdx.x;
  const int n = i / K, k = i % K;
  d[i] = (__bf16)s[(size_t)k * N + n];
}

// ---------------------------------------------------------------------------
// GEMM 1: qkv = x @ w_attn + b_attn, scatter into q/k/v [B,H,S,64] (q scaled)
// block tile 64x256, 8 waves of 32x64, K-step 32,
// double-buffered LDS fed by async global->LDS copies (software pipeline)
// ---------------------------------------------------------------------------
__global__ __launch_bounds__(256) void k_gemm_qkv(
    const __bf16* __restrict__ A,    // [8192][1024] x (bf16)
    const __bf16* __restrict__ BT,   // [3072][1024] w_attn^T (bf16)
    const float* __restrict__ bias,  // [3072]
    __bf16* __restrict__ qo, __bf16* __restrict__ ko, __bf16* __restrict__ vo) {
  constexpr int K = 1024;
  __shared__ unsigned short sA[2][64][40];
  __shared__ unsigned short sB[2][256][40];
  const int tid = threadIdx.x, lane = tid & 31, wave = tid >> 5;
  const int hi = lane >> 4, ln = lane & 15;
  const int m0 = blockIdx.y * 64, n0 = blockIdx.x * 256;
  const int wm = (wave & 1) * 32, wn = (wave >> 1) * 64;

  const int ra = tid >> 2, ca = (tid & 3) * 8;  // A-tile staging coords
  auto stage = [&](int k0, int buf) {
    async_copy_b128(lds_off(&sA[buf][ra][ca]),
                    &A[(size_t)(m0 + ra) * K + k0 + ca]);
    const unsigned lb = lds_off(&sB[buf][tid][0]);
    const unsigned long long gb =
        (unsigned long long)&BT[(size_t)(n0 + tid) * K + k0];
#pragma unroll
    for (int cc = 0; cc < 4; ++cc)
      asm volatile("global_load_async_to_lds_b128 %0, %1, off"
                   :: "v"(lb + 16u * cc), "v"(gb + 16u * cc) : "memory");
  };

  v8f acc[2][4];
#pragma unroll
  for (int i = 0; i < 2; ++i)
#pragma unroll
    for (int j = 0; j < 4; ++j) acc[i][j] = v8f_zero();

  stage(0, 0);
  for (int k0 = 0; k0 < K; k0 += 32) {
    const int buf = (k0 >> 5) & 1;
    wait_async0();     // this wave's deposits for `buf` complete
    __syncthreads();   // everyone's deposits visible; prior reads retired
    if (k0 + 32 < K) stage(k0 + 32, buf ^ 1);  // overlap next copy w/ compute
    if (k0 + 64 < K)
      __builtin_prefetch(&A[(size_t)(m0 + ra) * K + k0 + 64 + ca], 0, 1);

    v16bf aF[2], bF[4];
#pragma unroll
    for (int mt = 0; mt < 2; ++mt)
      aF[mt] = load_fragA_lds(sA[buf], wm + mt * 16 + ln, hi, 0);
#pragma unroll
    for (int nt = 0; nt < 4; ++nt)
      bF[nt] = load_fragB_lds(sB[buf], wn + nt * 16 + ln, hi, 0);
#pragma unroll
    for (int mt = 0; mt < 2; ++mt)
#pragma unroll
      for (int nt = 0; nt < 4; ++nt)
        acc[mt][nt] = wmma_bf16(aF[mt], bF[nt], acc[mt][nt]);
  }

#pragma unroll
  for (int mt = 0; mt < 2; ++mt)
#pragma unroll
    for (int nt = 0; nt < 4; ++nt) {
      const int n = n0 + wn + nt * 16 + ln;
      const float bv = bias[n];
      const int which = n >> 10, n1 = n & 1023;
      const int hh = n1 >> 6, dd = n1 & 63;
      __bf16* dst = (which == 0) ? qo : (which == 1) ? ko : vo;
      const float mul = (which == 0) ? 0.125f : 1.0f;  // fold 1/sqrt(64) into q
#pragma unroll
      for (int r = 0; r < 8; ++r) {
        const int m = m0 + wm + mt * 16 + r + 8 * hi;  // C layout: row = r + 8*hi
        const int bb = m >> 11, ss = m & 2047;
        dst[((size_t)((bb * 16 + hh) * 2048 + ss) << 6) + dd] =
            (__bf16)((acc[mt][nt][r] + bv) * mul);
      }
    }
}

// ---------------------------------------------------------------------------
// Flash attention: one workgroup = 128 query rows of one (b,h); wave = 16 rows
// K blocks staged with async global->LDS copies; V transposed manually
// ---------------------------------------------------------------------------
__global__ __launch_bounds__(256) void k_attn(
    const __bf16* __restrict__ qg, const __bf16* __restrict__ kg,
    const __bf16* __restrict__ vg, __bf16* __restrict__ og) {
  constexpr int S = 2048, HD = 64;
  __shared__ unsigned short sK[32][72];      // K block, row-major [key][d]
  __shared__ unsigned short sVT[64][40];     // V block transposed [d][key]
  __shared__ unsigned short sP[8][16][40];   // per-wave P tile [m][key]
  const int tid = threadIdx.x, lane = tid & 31, wave = tid >> 5;
  const int hi = lane >> 4, ln = lane & 15;
  const int bh = blockIdx.x;                 // b*16 + h
  const int b = bh >> 4, h = bh & 15;
  const int q0wg = blockIdx.y * 128;
  const int q0 = q0wg + wave * 16;
  const __bf16* qb = qg + (size_t)bh * S * HD;
  const __bf16* kbp = kg + (size_t)bh * S * HD;
  const __bf16* vbp = vg + (size_t)bh * S * HD;

  v16bf aQ[2];  // Q tile 16x64 as two 16x32 A fragments (pre-scaled by 1/8)
#pragma unroll
  for (int c = 0; c < 2; ++c) aQ[c] = load_fragA_glb(qb, HD, q0 + ln, hi, c * 32);

  float rmax[8], rsum[8];
  v8f accO[4];
#pragma unroll
  for (int r = 0; r < 8; ++r) { rmax[r] = -1e30f; rsum[r] = 0.0f; }
#pragma unroll
  for (int c = 0; c < 4; ++c) accO[c] = v8f_zero();

  const int kend = q0wg + 128;  // causal bound for the whole workgroup
  for (int kblk = 0; kblk < kend; kblk += 32) {
    {  // stage K via async copy; transpose-scatter V via DS
      const int r = tid >> 3, c = (tid & 7) * 8;
      async_copy_b128(lds_off(&sK[r][c]), &kbp[(size_t)(kblk + r) * HD + c]);
      const uint4 vv = *(const uint4*)&vbp[(size_t)(kblk + r) * HD + c];
      const unsigned short* vs = (const unsigned short*)&vv;
#pragma unroll
      for (int j = 0; j < 8; ++j) sVT[c + j][r] = vs[j];
      wait_async0();
    }
    __syncthreads();

    // scores S = Q @ K^T (two 16-key tiles, chained over hd chunks)
    v8f sc[2];
#pragma unroll
    for (int nt = 0; nt < 2; ++nt) {
      v8f cf = v8f_zero();
#pragma unroll
      for (int c = 0; c < 2; ++c) {
        v16bf bK = load_fragB_lds(sK, nt * 16 + ln, hi, c * 32);
        cf = wmma_bf16(aQ[c], bK, cf);
      }
      sc[nt] = cf;
    }

    // online softmax (per row r: row = q0 + r + 8*hi; col = lane&15)
#pragma unroll
    for (int r = 0; r < 8; ++r) {
      const int qi = q0 + r + 8 * hi;
#pragma unroll
      for (int nt = 0; nt < 2; ++nt) {
        const int ki = kblk + nt * 16 + ln;
        if (ki > qi) sc[nt][r] = -1e30f;  // causal mask (select, no branch)
      }
      float mx = fmaxf(sc[0][r], sc[1][r]);
#pragma unroll
      for (int o = 8; o >= 1; o >>= 1) mx = fmaxf(mx, __shfl_xor(mx, o, 32));
      const float nm = fmaxf(rmax[r], mx);
      const float scale = __expf(rmax[r] - nm);
      rmax[r] = nm;
      const float p0 = __expf(sc[0][r] - nm);
      const float p1 = __expf(sc[1][r] - nm);
      sc[0][r] = p0;
      sc[1][r] = p1;
      float ps = p0 + p1;
#pragma unroll
      for (int o = 8; o >= 1; o >>= 1) ps += __shfl_xor(ps, o, 32);
      rsum[r] = rsum[r] * scale + ps;
#pragma unroll
      for (int c = 0; c < 4; ++c) accO[c][r] *= scale;
    }

    // reshape P: C-layout -> LDS -> A-fragment (per-wave private tile)
#pragma unroll
    for (int nt = 0; nt < 2; ++nt)
#pragma unroll
      for (int r = 0; r < 8; ++r)
        sP[wave][r + 8 * hi][nt * 16 + ln] = bf16_bits(sc[nt][r]);
    asm volatile("s_wait_dscnt 0" ::: "memory");  // own-wave DS RAW fence

    v16bf aP = load_fragA_lds(sP[wave], ln, hi, 0);
#pragma unroll
    for (int c = 0; c < 4; ++c) {
      v16bf bV = load_fragB_lds(sVT, c * 16 + ln, hi, 0);
      accO[c] = wmma_bf16(aP, bV, accO[c]);
    }
    __syncthreads();
  }

  // normalize and store merged-head output [B*S][1024] (bf16 staging)
#pragma unroll
  for (int c = 0; c < 4; ++c) {
    const int d = h * 64 + c * 16 + ln;
#pragma unroll
    for (int r = 0; r < 8; ++r) {
      const int qi = q0 + r + 8 * hi;
      og[(size_t)(b * S + qi) * 1024 + d] = (__bf16)(accO[c][r] / rsum[r]);
    }
  }
}

// ---------------------------------------------------------------------------
// GEMM 2: out = attn_out @ w_proj + b_proj (fp32 output), same async pipeline
// ---------------------------------------------------------------------------
__global__ __launch_bounds__(256) void k_gemm_proj(
    const __bf16* __restrict__ A,    // [8192][1024] attn out (bf16)
    const __bf16* __restrict__ BT,   // [1024][1024] w_proj^T (bf16)
    const float* __restrict__ bias,  // [1024]
    float* __restrict__ out) {       // [8192][1024] fp32
  constexpr int K = 1024;
  __shared__ unsigned short sA[2][64][40];
  __shared__ unsigned short sB[2][256][40];
  const int tid = threadIdx.x, lane = tid & 31, wave = tid >> 5;
  const int hi = lane >> 4, ln = lane & 15;
  const int m0 = blockIdx.y * 64, n0 = blockIdx.x * 256;
  const int wm = (wave & 1) * 32, wn = (wave >> 1) * 64;

  const int ra = tid >> 2, ca = (tid & 3) * 8;
  auto stage = [&](int k0, int buf) {
    async_copy_b128(lds_off(&sA[buf][ra][ca]),
                    &A[(size_t)(m0 + ra) * K + k0 + ca]);
    const unsigned lb = lds_off(&sB[buf][tid][0]);
    const unsigned long long gb =
        (unsigned long long)&BT[(size_t)(n0 + tid) * K + k0];
#pragma unroll
    for (int cc = 0; cc < 4; ++cc)
      asm volatile("global_load_async_to_lds_b128 %0, %1, off"
                   :: "v"(lb + 16u * cc), "v"(gb + 16u * cc) : "memory");
  };

  v8f acc[2][4];
#pragma unroll
  for (int i = 0; i < 2; ++i)
#pragma unroll
    for (int j = 0; j < 4; ++j) acc[i][j] = v8f_zero();

  stage(0, 0);
  for (int k0 = 0; k0 < K; k0 += 32) {
    const int buf = (k0 >> 5) & 1;
    wait_async0();
    __syncthreads();
    if (k0 + 32 < K) stage(k0 + 32, buf ^ 1);
    if (k0 + 64 < K)
      __builtin_prefetch(&A[(size_t)(m0 + ra) * K + k0 + 64 + ca], 0, 1);

    v16bf aF[2], bF[4];
#pragma unroll
    for (int mt = 0; mt < 2; ++mt)
      aF[mt] = load_fragA_lds(sA[buf], wm + mt * 16 + ln, hi, 0);
#pragma unroll
    for (int nt = 0; nt < 4; ++nt)
      bF[nt] = load_fragB_lds(sB[buf], wn + nt * 16 + ln, hi, 0);
#pragma unroll
    for (int mt = 0; mt < 2; ++mt)
#pragma unroll
      for (int nt = 0; nt < 4; ++nt)
        acc[mt][nt] = wmma_bf16(aF[mt], bF[nt], acc[mt][nt]);
  }

#pragma unroll
  for (int mt = 0; mt < 2; ++mt)
#pragma unroll
    for (int nt = 0; nt < 4; ++nt) {
      const int n = n0 + wn + nt * 16 + ln;
      const float bv = bias[n];
#pragma unroll
      for (int r = 0; r < 8; ++r) {
        const int m = m0 + wm + mt * 16 + r + 8 * hi;
        out[(size_t)m * 1024 + n] = acc[mt][nt][r] + bv;
      }
    }
}

// ---------------------------------------------------------------------------
// Launcher
// ---------------------------------------------------------------------------
extern "C" void kernel_launch(void* const* d_in, const int* in_sizes, int n_in,
                              void* d_out, int out_size, void* d_ws,
                              size_t ws_size, hipStream_t stream) {
  (void)in_sizes; (void)n_in; (void)out_size; (void)ws_size;
  const float* x      = (const float*)d_in[0];  // [4,2048,1024]
  const float* w_attn = (const float*)d_in[1];  // [1024,3072]
  const float* b_attn = (const float*)d_in[2];  // [3072]
  const float* w_proj = (const float*)d_in[3];  // [1024,1024]
  const float* b_proj = (const float*)d_in[4];  // [1024]
  float* out = (float*)d_out;                   // [4,2048,1024]

  const size_t T = (size_t)8 * 1024 * 1024;  // 8,388,608 = B*S*D = B*H*S*hd
  __bf16* ws  = (__bf16*)d_ws;
  __bf16* qw  = ws;                 // [B,H,S,64] (pre-scaled)
  __bf16* kw  = qw + T;             // [B,H,S,64]
  __bf16* vw  = kw + T;             // [B,H,S,64]
  __bf16* xb  = vw + T;             // [8192,1024]
  __bf16* wAT = xb + T;             // [3072,1024] = w_attn^T
  __bf16* wPT = wAT + (size_t)3072 * 1024;  // [1024,1024] = w_proj^T
  __bf16* ao  = wPT + (size_t)1024 * 1024;  // [8192,1024] merged heads

  k_cast_bf16<<<(int)(T / 256), 256, 0, stream>>>(x, xb);
  k_transpose_bf16<<<(3072 * 1024) / 256, 256, 0, stream>>>(w_attn, wAT, 1024, 3072);
  k_transpose_bf16<<<(1024 * 1024) / 256, 256, 0, stream>>>(w_proj, wPT, 1024, 1024);
  k_gemm_qkv<<<dim3(12, 128), 256, 0, stream>>>(xb, wAT, b_attn, qw, kw, vw);
  k_attn<<<dim3(64, 16), 256, 0, stream>>>(qw, kw, vw, ao);
  k_gemm_proj<<<dim3(4, 128), 256, 0, stream>>>(ao, wPT, b_proj, out);
}
